// infect_nn_lstm_59536836657604
// MI455X (gfx1250) — matrix operand order repaired and checked by reading, not compiled, not run on hardware
//
#include <hip/hip_runtime.h>

typedef _Float16 v16h __attribute__((ext_vector_type(16)));
typedef float    v8f  __attribute__((ext_vector_type(8)));
typedef float    v2f  __attribute__((ext_vector_type(2)));

union V16H {
    v16h     v;
    _Float16 h[16];
    unsigned u[8];
};

#if defined(__has_builtin)
#if __has_builtin(__builtin_amdgcn_tanhf)
#define HW_TANH 1
#endif
#endif

static __device__ __forceinline__ v8f wmma_f16(v16h a, v16h b, v8f c) {
    // D = A(16x32 f16) * B(32x16 f16) + C(f32)
    return __builtin_amdgcn_wmma_f32_16x16x32_f16(false, a, false, b, (short)0, c, false, false);
}

static __device__ __forceinline__ float tanh_f(float x) {
#ifdef HW_TANH
    return __builtin_amdgcn_tanhf(x);           // v_tanh_f32: 1 TRANS op
#else
    return 1.0f - 2.0f * __builtin_amdgcn_rcpf(__expf(2.0f * x) + 1.0f);
#endif
}
static __device__ __forceinline__ v2f tanh2(v2f x) {
    v2f r; r[0] = tanh_f(x[0]); r[1] = tanh_f(x[1]); return r;
}
static __device__ __forceinline__ unsigned pk2(float a, float b) {
    return __builtin_bit_cast(unsigned, __builtin_amdgcn_cvt_pkrtz(a, b));
}
// lane <-> lane^16 swap: ds_swizzle group-of-32, xor_mask=0x10, and_mask=0x1f
static __device__ __forceinline__ unsigned swap16(unsigned x) {
    return (unsigned)__builtin_amdgcn_ds_swizzle((int)x, 0x401f);
}
static __device__ __forceinline__ float swap16f(float x) {
    return __builtin_bit_cast(float,
        (unsigned)__builtin_amdgcn_ds_swizzle((int)__builtin_bit_cast(unsigned, x), 0x401f));
}

// One wave32 handles a 16-sample tile.
// Orientation: D[m,n] = sum_k A[m,k]*B[k,n], with n = batch sample, m = unit row.
// C/D layout: lane n (0..15) holds rows m=0..7 in acc[0..7]; lane n+16 holds m=8..15.
// B layout:   lane n holds K=0..7 & 16..23 (2 per dword); lane n+16 holds K=8..15 & 24..31.
// A layout:   lane L holds row L%16; lanes 0-15: K=0..7 & 16..23; lanes 16-31: K=8..15 & 24..31.
// Scaling tricks:
//   - i/f/o gate rows pre-scaled by 0.5 so sigmoid(z) = 0.5*tanh(z/2)+0.5 needs no input mul.
//   - the hidden state is carried as H = 2h (one FMA: H = to*tc + tc); all A-operand
//     h-columns (K=0..15) carry an extra 0.5 to compensate.
__global__ __launch_bounds__(256) void infect_lstm_wmma_kernel(
    const float* __restrict__ mobility, const float* __restrict__ controls,
    const float* __restrict__ W_cc, const float* __restrict__ b_cc,
    const float* __restrict__ W_ih, const float* __restrict__ W_hh,
    const float* __restrict__ b_ih, const float* __restrict__ b_hh,
    const float* __restrict__ W1, const float* __restrict__ b1,
    const float* __restrict__ W2, const float* __restrict__ b2,
    const float* __restrict__ W3, const float* __restrict__ b3,
    float* __restrict__ out, int nB)
{
    const int tid   = threadIdx.x;
    const int lane  = tid & 31;
    const int wib   = tid >> 5;
    const int half  = lane >> 4;          // 0: low half, 1: high half
    const int col   = lane & 15;          // batch column inside the 16-wide tile
    const long base = ((long)blockIdx.x * 8 + wib) * 16;
    if (base >= (long)nB) return;

    const long sReq  = base + col;
    const bool valid = sReq < (long)nB;
    const long s     = valid ? sReq : (long)nB - 1;

    const int mrow = lane & 15;                       // A row this lane supplies
    const int k0   = half ? 8 : 0;                    // A K-offset for this lane half
    const unsigned lomask = half ? 0u : 0xFFFFFFFFu;  // keep B dwords only in low half

    // ---------------- A operands (constant: built once per thread) ----------------
    // LSTM recurrent weights; K=16 -> W_ih, K=17 -> bias.
    V16H A_g[4];
#pragma unroll
    for (int g = 0; g < 4; ++g) {
        const float scg = (g == 2) ? 1.0f : 0.5f;     // sigmoid-gate pre-scale
        const float sch = 0.5f * scg;                 // + compensate H = 2h on h-columns
        const float4* wr = (const float4*)(W_hh + (g * 16 + mrow) * 16 + k0);  // 32B aligned
        const float4 wa = wr[0], wb = wr[1];
        A_g[g].u[0] = pk2(wa.x * sch, wa.y * sch);
        A_g[g].u[1] = pk2(wa.z * sch, wa.w * sch);
        A_g[g].u[2] = pk2(wb.x * sch, wb.y * sch);
        A_g[g].u[3] = pk2(wb.z * sch, wb.w * sch);
        const unsigned xb = pk2(W_ih[g * 16 + mrow] * scg,
                                (b_ih[g * 16 + mrow] + b_hh[g * 16 + mrow]) * scg);
        A_g[g].u[4] = half ? 0u : xb;                 // K=16: x coeff, K=17: bias
        A_g[g].u[5] = 0u; A_g[g].u[6] = 0u; A_g[g].u[7] = 0u;
    }

    // FC1: W1 is [16,19]; K=16..18 -> control features, K=19 -> bias column.
    V16H A1;
    {
        float w[8];
#pragma unroll
        for (int j = 0; j < 8; ++j) w[j] = W1[mrow * 19 + k0 + j] * 0.5f;  // H = 2h comp.
        A1.u[0] = pk2(w[0], w[1]);
        A1.u[1] = pk2(w[2], w[3]);
        A1.u[2] = pk2(w[4], w[5]);
        A1.u[3] = pk2(w[6], w[7]);
        A1.u[4] = half ? 0u : pk2(W1[mrow * 19 + 16], W1[mrow * 19 + 17]);
        A1.u[5] = half ? 0u : pk2(W1[mrow * 19 + 18], b1[mrow]);   // K=18, K=19: bias
        A1.u[6] = 0u; A1.u[7] = 0u;
    }

    // FC2: W2 is [16,16]; K=16 -> bias column. (y1 is stored plain, no comp. needed.)
    V16H A2;
    {
        const float4* wr = (const float4*)(W2 + mrow * 16 + k0);   // 32B aligned
        const float4 wa = wr[0], wb = wr[1];
        A2.u[0] = pk2(wa.x, wa.y);
        A2.u[1] = pk2(wa.z, wa.w);
        A2.u[2] = pk2(wb.x, wb.y);
        A2.u[3] = pk2(wb.z, wb.w);
        A2.u[4] = half ? 0u : pk2(b2[mrow], 0.0f);
        A2.u[5] = 0u; A2.u[6] = 0u; A2.u[7] = 0u;
    }

    v2f w3p[4];
#pragma unroll
    for (int r = 0; r < 4; ++r) {
        w3p[r][0] = W3[half * 8 + 2 * r];
        w3p[r][1] = W3[half * 8 + 2 * r + 1];
    }
    const float bias3 = b3[0];

    // ---------------- per-sample inputs ----------------
    unsigned xpack[7];
#pragma unroll
    for (int t = 0; t < 7; ++t) xpack[t] = pk2(mobility[s * 7 + t], 1.0f);

    float cin[5];
#pragma unroll
    for (int j = 0; j < 5; ++j) cin[j] = controls[s * 5 + j];

    float ctrl[3];
#pragma unroll
    for (int k = 0; k < 3; ++k) {
        float acc = b_cc[k];
#pragma unroll
        for (int j = 0; j < 5; ++j) acc = fmaf(W_cc[k * 5 + j], cin[j], acc);
        ctrl[k] = acc;
    }

    // ---------------- LSTM: 7 steps ----------------
    const v8f zero = {};
    const v2f vhalf = {0.5f, 0.5f};
    v2f cst[4];
    unsigned p[4];                          // this half's 8 H-values, f16x2-packed

#pragma unroll
    for (int t = 0; t < 7; ++t) {
        V16H B;
        if (t == 0) {
            // h == 0: only the x/bias column is live (low half, K=16..17).
#pragma unroll
            for (int i = 0; i < 8; ++i) B.u[i] = 0u;
            B.u[4] = xpack[0] & lomask;
        } else {
            unsigned q[4];
#pragma unroll
            for (int i = 0; i < 4; ++i) q[i] = swap16(p[i]);
            B.u[0] = p[0];
            B.u[1] = p[1];
            B.u[2] = p[2];
            B.u[3] = p[3];
            B.u[4] = xpack[t] & lomask;                // K=16: x_t, K=17: 1.0 (low half)
            B.u[5] = 0u; B.u[6] = 0u; B.u[7] = 0u;
        }
        v8f gi = wmma_f16(A_g[0].v, B.v, zero);
        v8f gf = wmma_f16(A_g[1].v, B.v, zero);    // dead at t==0 (c==0) -> DCE
        v8f gg = wmma_f16(A_g[2].v, B.v, zero);
        v8f go = wmma_f16(A_g[3].v, B.v, zero);

#pragma unroll
        for (int rp = 0; rp < 4; ++rp) {
            v2f gi2 = {gi[2 * rp], gi[2 * rp + 1]};
            v2f gg2 = {gg[2 * rp], gg[2 * rp + 1]};
            v2f go2 = {go[2 * rp], go[2 * rp + 1]};
            v2f ti = tanh2(gi2);
            v2f tg = tanh2(gg2);
            v2f to = tanh2(go2);
            v2f iv = __builtin_elementwise_fma(ti, vhalf, vhalf);   // sigmoid(i)
            v2f cv;
            if (t == 0) {
                cv = iv * tg;
            } else {
                v2f gf2 = {gf[2 * rp], gf[2 * rp + 1]};
                v2f fv = __builtin_elementwise_fma(tanh2(gf2), vhalf, vhalf);
                cv = __builtin_elementwise_fma(fv, cst[rp], iv * tg);
            }
            cst[rp] = cv;
            v2f tc = tanh2(cv);
            v2f H  = __builtin_elementwise_fma(to, tc, tc);         // H = 2*o*tanh(c)
            p[rp] = pk2(H[0], H[1]);
        }
    }

    // ---------------- FC1: concat(h, ctrl) -> 16, tanh ----------------
    {
        const unsigned c01 = pk2(ctrl[0], ctrl[1]);    // K=16,17
        const unsigned c2b = pk2(ctrl[2], 1.0f);       // K=18, K=19: bias lane
        unsigned q[4];
#pragma unroll
        for (int i = 0; i < 4; ++i) q[i] = swap16(p[i]);
        V16H B;
        B.u[0] = p[0];
        B.u[1] = p[1];
        B.u[2] = p[2];
        B.u[3] = p[3];
        B.u[4] = c01 & lomask;
        B.u[5] = c2b & lomask;
        B.u[6] = 0u; B.u[7] = 0u;
        v8f a1 = wmma_f16(A1.v, B.v, zero);
#pragma unroll
        for (int rp = 0; rp < 4; ++rp) {
            v2f a2v = {a1[2 * rp], a1[2 * rp + 1]};
            v2f y = tanh2(a2v);
            p[rp] = pk2(y[0], y[1]);
        }
    }

    // ---------------- FC2: 16 -> 16, tanh ----------------
    v2f y2[4];
    {
        unsigned q[4];
#pragma unroll
        for (int i = 0; i < 4; ++i) q[i] = swap16(p[i]);
        V16H B;
        B.u[0] = p[0];
        B.u[1] = p[1];
        B.u[2] = p[2];
        B.u[3] = p[3];
        B.u[4] = half ? 0u : 0x00003c00u;              // K=16: bias lane (1.0f16)
        B.u[5] = 0u; B.u[6] = 0u; B.u[7] = 0u;
        v8f a2 = wmma_f16(A2.v, B.v, zero);
#pragma unroll
        for (int rp = 0; rp < 4; ++rp) {
            v2f av = {a2[2 * rp], a2[2 * rp + 1]};
            y2[rp] = tanh2(av);
        }
    }

    // ---------------- FC3: 16 -> 1, ReLU ----------------
    v2f acc2 = {0.0f, 0.0f};
#pragma unroll
    for (int rp = 0; rp < 4; ++rp)
        acc2 = __builtin_elementwise_fma(w3p[rp], y2[rp], acc2);
    const float part = acc2[0] + acc2[1];
    const float tot  = part + swap16f(part);
    const float val  = fmaxf(tot + bias3, 0.0f);
    if ((lane < 16) && valid) out[s] = val;
}

extern "C" void kernel_launch(void* const* d_in, const int* in_sizes, int n_in,
                              void* d_out, int out_size, void* d_ws, size_t ws_size,
                              hipStream_t stream) {
    (void)n_in; (void)d_ws; (void)ws_size; (void)out_size;
    const float* mobility = (const float*)d_in[0];
    const float* controls = (const float*)d_in[1];
    // d_in[2] = "last" is unused by the reference graph
    const float* W_cc = (const float*)d_in[3];
    const float* b_cc = (const float*)d_in[4];
    const float* W_ih = (const float*)d_in[5];
    const float* W_hh = (const float*)d_in[6];
    const float* b_ih = (const float*)d_in[7];
    const float* b_hh = (const float*)d_in[8];
    const float* W1   = (const float*)d_in[9];
    const float* b1   = (const float*)d_in[10];
    const float* W2   = (const float*)d_in[11];
    const float* b2   = (const float*)d_in[12];
    const float* W3   = (const float*)d_in[13];
    const float* b3   = (const float*)d_in[14];
    float* out = (float*)d_out;

    const int nB = in_sizes[2];                 // "last" has exactly B elements
    const int samplesPerBlock = 128;            // 8 waves * 16 samples
    const int blocks = (nB + samplesPerBlock - 1) / samplesPerBlock;

    infect_lstm_wmma_kernel<<<blocks, 256, 0, stream>>>(
        mobility, controls, W_cc, b_cc, W_ih, W_hh, b_ih, b_hh,
        W1, b1, W2, b2, W3, b3, out, nB);
}
